// MultiHeadAttention_2388001816932
// MI455X (gfx1250) — matrix operand order repaired and checked by reading, not compiled, not run on hardware
//
#include <hip/hip_runtime.h>
#include <hip/hip_bf16.h>

// ---------------------------------------------------------------------------
// MultiHeadAttention (B=8,S=1024,H=16,D=48) for MI455X / gfx1250, wave32 WMMA
// bf16 matmuls with fp32 accumulation via v_wmma_f32_16x16x32_bf16.
// Column-softmax (softmax over the QUERY axis) via online (m,l) per column.
// Shared tiles staged in LDS with async global->LDS copies when available.
// ---------------------------------------------------------------------------

typedef __bf16 bf16;
typedef __attribute__((ext_vector_type(16))) __bf16 bf16x16;
typedef __attribute__((ext_vector_type(8)))  float  f32x8;
typedef __attribute__((ext_vector_type(4)))  int    i32x4;

#define DIMK   768
#define SEQ    1024
#define NH     16
#define HD     48
#define HDP    64      // padded head dim for WMMA K-loop (2 x 32)
#define BH     128     // B*H
#define NROWS  8192    // B*S

static constexpr float kInvSqrtD = 0.14433756729740643f;  // 1/sqrt(48)

#if __has_builtin(__builtin_amdgcn_global_load_async_to_lds_b128)
#define HAVE_ASYNC_LDS 1
#else
#define HAVE_ASYNC_LDS 0
#endif

// Copy 16 bytes global -> LDS (async-to-LDS if the toolchain exposes it).
static __device__ __forceinline__ void stage16(const bf16* g, bf16* l) {
#if HAVE_ASYNC_LDS
  __builtin_amdgcn_global_load_async_to_lds_b128(
      (__attribute__((address_space(1))) i32x4*)g,
      (__attribute__((address_space(3))) i32x4*)l, 0, 0);
#else
  *reinterpret_cast<uint4*>(l) = *reinterpret_cast<const uint4*>(g);
#endif
}

static __device__ __forceinline__ void stage_wait() {
#if HAVE_ASYNC_LDS
#if __has_builtin(__builtin_amdgcn_s_wait_asynccnt)
  __builtin_amdgcn_s_wait_asynccnt(0);
#else
  asm volatile("s_wait_asynccnt 0x0" ::: "memory");
#endif
#endif
}

// Load a 16-element bf16 fragment from two 16-byte chunks (global or LDS).
static __device__ __forceinline__ bf16x16 ld_frag(const bf16* p0, const bf16* p1) {
  union { uint4 u[2]; bf16x16 v; } t;
  t.u[0] = *reinterpret_cast<const uint4*>(p0);
  t.u[1] = *reinterpret_cast<const uint4*>(p1);
  return t.v;
}

static __device__ __forceinline__ f32x8 wmma_bf16(bf16x16 a, bf16x16 b, f32x8 c) {
  // (neg_a, A, neg_b, B, c_mod, C, reuse_a, reuse_b)
  return __builtin_amdgcn_wmma_f32_16x16x32_bf16(false, a, false, b, (short)0, c,
                                                 false, false);
}

// ---------------------------------------------------------------------------
// Prep kernels
// ---------------------------------------------------------------------------
__global__ __launch_bounds__(256) void zero_u4_kernel(uint4* __restrict__ p, int n) {
  int i = blockIdx.x * 256 + threadIdx.x;
  if (i < n) p[i] = make_uint4(0u, 0u, 0u, 0u);
}

__global__ __launch_bounds__(256) void cvt_x_kernel(const float* __restrict__ x,
                                                    bf16* __restrict__ xb, int n4) {
  int i = blockIdx.x * 256 + threadIdx.x;
  if (i >= n4) return;
  float4 v = reinterpret_cast<const float4*>(x)[i];
  union { bf16 b[4]; uint2 u; } t;
  t.b[0] = (bf16)v.x; t.b[1] = (bf16)v.y; t.b[2] = (bf16)v.z; t.b[3] = (bf16)v.w;
  reinterpret_cast<uint2*>(xb)[i] = t.u;
}

// Wq[h][m][d] (fp32) -> Wt[(h*48+d)][m] (bf16), i.e. B^T layout [N][K].
__global__ __launch_bounds__(256) void build_wht_kernel(const float* __restrict__ W,
                                                        bf16* __restrict__ Wt, int n) {
  int i = blockIdx.x * 256 + threadIdx.x;
  if (i >= n) return;
  int h   = i / (DIMK * HD);
  int rem = i - h * (DIMK * HD);
  int m   = rem / HD;
  int d   = rem - m * HD;
  Wt[(size_t)(h * HD + d) * DIMK + m] = (bf16)W[i];
}

// Wo[m][n] (fp32) -> Wot[n][m] (bf16).
__global__ __launch_bounds__(256) void build_wot_kernel(const float* __restrict__ W,
                                                        bf16* __restrict__ Wt, int n) {
  int i = blockIdx.x * 256 + threadIdx.x;
  if (i >= n) return;
  int m = i / DIMK;
  int c = i - m * DIMK;
  Wt[(size_t)c * DIMK + m] = (bf16)W[i];
}

// ---------------------------------------------------------------------------
// WMMA GEMM: C[8192 x 768] = A[8192 x 768] * B, B given transposed [N][K].
// Each wave computes a 32(M) x 64(N) tile: 8 WMMA per 32-deep K step.
// mode 0/1: store bf16 to padded [BH][S][64] layout (Q scaled by 1/sqrt(48))
// mode 2  : store bf16 to transposed [BH][48][S] layout (V)
// mode 3  : store fp32 row-major [8192][768] (+bias) -> final output
// ---------------------------------------------------------------------------
__global__ __launch_bounds__(256) void gemm_kernel(const bf16* __restrict__ A,
                                                   const bf16* __restrict__ Bt,
                                                   const float* __restrict__ bias,
                                                   bf16* __restrict__ ob,
                                                   float* __restrict__ of,
                                                   int mode, float scale) {
  const int wid  = blockIdx.x * 8 + (threadIdx.x >> 5);
  const int lane = threadIdx.x & 31;
  const int l15  = lane & 15;
  const int hi   = lane >> 4;
  const int mt   = wid & 255;   // 256 M-tiles of 32 rows
  const int ng   = wid >> 8;    // 12 N-groups of 64
  const int Mbase = mt << 5;
  const int Nbase = ng << 6;
  const int klo2 = hi << 3;     // unused (A and B both use klo)
  const int klo  = hi << 4;     // B-fragment chunk select (0/16)

  const bf16* ar0 = A + (size_t)(Mbase + l15) * DIMK;
  const bf16* ar1 = ar0 + (size_t)16 * DIMK;
  const bf16* br0 = Bt + (size_t)(Nbase +  0 + l15) * DIMK;
  const bf16* br1 = Bt + (size_t)(Nbase + 16 + l15) * DIMK;
  const bf16* br2 = Bt + (size_t)(Nbase + 32 + l15) * DIMK;
  const bf16* br3 = Bt + (size_t)(Nbase + 48 + l15) * DIMK;

  f32x8 c[2][4] = {{f32x8{}, f32x8{}, f32x8{}, f32x8{}},
                   {f32x8{}, f32x8{}, f32x8{}, f32x8{}}};
#pragma unroll 2
  for (int k = 0; k < DIMK; k += 32) {
    __builtin_prefetch(ar0 + k + 64, 0, 3);  // global_prefetch_b8
    __builtin_prefetch(ar1 + k + 64, 0, 3);
    bf16x16 a0 = ld_frag(ar0 + k + klo, ar0 + k + klo + 8);
    bf16x16 a1 = ld_frag(ar1 + k + klo, ar1 + k + klo + 8);
    bf16x16 b0 = ld_frag(br0 + k + klo, br0 + k + klo + 8);
    bf16x16 b1 = ld_frag(br1 + k + klo, br1 + k + klo + 8);
    bf16x16 b2 = ld_frag(br2 + k + klo, br2 + k + klo + 8);
    bf16x16 b3 = ld_frag(br3 + k + klo, br3 + k + klo + 8);
    c[0][0] = wmma_bf16(a0, b0, c[0][0]);
    c[1][0] = wmma_bf16(a1, b0, c[1][0]);
    c[0][1] = wmma_bf16(a0, b1, c[0][1]);
    c[1][1] = wmma_bf16(a1, b1, c[1][1]);
    c[0][2] = wmma_bf16(a0, b2, c[0][2]);
    c[1][2] = wmma_bf16(a1, b2, c[1][2]);
    c[0][3] = wmma_bf16(a0, b3, c[0][3]);
    c[1][3] = wmma_bf16(a1, b3, c[1][3]);
  }

#pragma unroll
  for (int t2 = 0; t2 < 2; ++t2) {
    const int Mb = Mbase + t2 * 16;
    const int b  = Mb >> 10;                      // batch index
    const int s0 = (Mb & 1023) + (hi << 3);       // sequence row base
#pragma unroll
    for (int j = 0; j < 4; ++j) {
      const int col = Nbase + j * 16 + l15;
      const float bj = bias[col];
      if (mode == 3) {
#pragma unroll
        for (int r = 0; r < 8; ++r) {
          int row = Mb + r + (hi << 3);
          of[(size_t)row * DIMK + col] = c[t2][j][r] + bj;
        }
      } else {
        const int h = col / HD;
        const int d = col - h * HD;
        if (mode == 2) {   // V transposed: vt[(bh*48 + d)*1024 + s]
          bf16* dst = ob + ((size_t)(b * NH + h) * HD + d) * SEQ;
#pragma unroll
          for (int r = 0; r < 8; ++r) dst[s0 + r] = (bf16)(c[t2][j][r] + bj);
        } else {           // Q/K padded: qk[(bh*1024 + s)*64 + d]
          bf16* dst = ob + (((size_t)(b * NH + h) * SEQ) << 6) + d;
#pragma unroll
          for (int r = 0; r < 8; ++r)
            dst[(size_t)(s0 + r) << 6] = (bf16)((c[t2][j][r] + bj) * scale);
        }
      }
    }
  }
}

// ---------------------------------------------------------------------------
// Column-softmax statistics (softmax over QUERY axis). Each wave owns 32
// key-columns of one (b,h) and streams all 1024 queries with online (m,l).
// The 16x64 Q tile of each step is staged once in LDS (async) and shared by
// all 8 waves. Key B-fragments are held in VGPRs for the whole loop.
// Grid: 128 bh * 4 kblocks (8 waves * 32 cols = 256 cols/WG), 256 threads.
// ---------------------------------------------------------------------------
__global__ __launch_bounds__(256) void softmax_stats_kernel(const bf16* __restrict__ qp,
                                                            const bf16* __restrict__ kp,
                                                            float* __restrict__ Mbuf,
                                                            float* __restrict__ Linv) {
  __shared__ __align__(16) bf16 qstage[16][HDP];   // 2 KB shared Q tile

  const int bh   = blockIdx.x >> 2;
  const int kblk = blockIdx.x & 3;
  const int wave = threadIdx.x >> 5;
  const int lane = threadIdx.x & 31;
  const int l15  = lane & 15;
  const int hi   = lane >> 4;
  const int klo  = hi << 4;
  const int klo2 = hi << 3;
  const int kbase = kblk * 256 + wave * 32;

  // Held B-fragments of K for two 16-column tiles ([bh][s][64] layout).
  const bf16* kr0 = kp + (((size_t)bh * SEQ + kbase + l15) << 6);
  const bf16* kr1 = kr0 + (16 << 6);
  bf16x16 kf00 = ld_frag(kr0 + klo,      kr0 + klo + 8);
  bf16x16 kf01 = ld_frag(kr0 + 32 + klo, kr0 + 32 + klo + 8);
  bf16x16 kf10 = ld_frag(kr1 + klo,      kr1 + klo + 8);
  bf16x16 kf11 = ld_frag(kr1 + 32 + klo, kr1 + 32 + klo + 8);

  const bf16* qb = qp + (((size_t)bh * SEQ) << 6);
  const int tid  = threadIdx.x;
  const int srow = tid >> 3;        // 0..31 (only <16 used)
  const int scol = (tid & 7) * 8;   // 16B chunk within 64-elem row

  float m0 = -3.0e38f, l0 = 0.0f, m1 = -3.0e38f, l1 = 0.0f;

  for (int qt = 0; qt < SEQ / 16; ++qt) {
    if (tid < 128)
      stage16(qb + ((size_t)(qt * 16 + srow) << 6) + scol, &qstage[srow][scol]);
    stage_wait();
    __syncthreads();

    bf16x16 a0 = ld_frag(&qstage[l15][klo],      &qstage[l15][klo + 8]);
    bf16x16 a1 = ld_frag(&qstage[l15][32 + klo], &qstage[l15][32 + klo + 8]);
    f32x8 s0 = {}, s1 = {};
    s0 = wmma_bf16(a0, kf00, s0); s0 = wmma_bf16(a1, kf01, s0);
    s1 = wmma_bf16(a0, kf10, s1); s1 = wmma_bf16(a1, kf11, s1);

#pragma unroll
    for (int t = 0; t < 2; ++t) {
      const f32x8& s = t ? s1 : s0;
      float& m = t ? m1 : m0;
      float& l = t ? l1 : l0;
      float tmax = s[0];
#pragma unroll
      for (int r = 1; r < 8; ++r) tmax = fmaxf(tmax, s[r]);
      tmax = fmaxf(tmax, __shfl_xor(tmax, 16, 32));   // merge column halves
      float mn = fmaxf(m, tmax);
      float ps = 0.0f;
#pragma unroll
      for (int r = 0; r < 8; ++r) ps += __expf(s[r] - mn);
      ps += __shfl_xor(ps, 16, 32);
      l = l * __expf(m - mn) + ps;
      m = mn;
    }
    __syncthreads();   // protect qstage before next overwrite
  }

  if (lane < 16) {
    size_t i0 = (size_t)bh * SEQ + kbase + lane;
    Mbuf[i0]      = m0;  Linv[i0]      = 1.0f / l0;
    Mbuf[i0 + 16] = m1;  Linv[i0 + 16] = 1.0f / l1;
  }
}

// ---------------------------------------------------------------------------
// PV pass: recompute score tiles, p = exp(s - m) * linv, transpose p through
// per-wave LDS staging (C-layout -> A-layout), accumulate attn[32q x 48d]
// over all 1024 keys. The 32-key K tile (4 KB) and V tile (3 KB) of each step
// are staged once in LDS (async) and shared by all 8 waves.
// Grid: 128 bh * 4 qblocks (8 waves * 32 q = 256 q/WG), 256 threads.
// ---------------------------------------------------------------------------
__global__ __launch_bounds__(256) void attn_pv_kernel(const bf16* __restrict__ qp,
                                                      const bf16* __restrict__ kp,
                                                      const bf16* __restrict__ vt,
                                                      const float* __restrict__ Mbuf,
                                                      const float* __restrict__ Linv,
                                                      bf16* __restrict__ attnC) {
  __shared__ __align__(16) bf16 kstage[32][HDP];     // 4 KB shared K tile
  __shared__ __align__(16) bf16 vstage[HD][32];      // 3 KB shared V tile
  __shared__ __align__(16) bf16 pstage[8][32][32];   // 16 KB per-wave p tiles

  const int bh   = blockIdx.x >> 2;
  const int qblk = blockIdx.x & 3;
  const int b    = bh >> 4;
  const int h    = bh & 15;
  const int wave = threadIdx.x >> 5;
  const int lane = threadIdx.x & 31;
  const int l15  = lane & 15;
  const int hi   = lane >> 4;
  const int hi8  = hi << 3;
  const int klo  = hi << 4;
  const int klo2 = hi << 3;
  const int qbase = qblk * 256 + wave * 32;

  // A-fragments of Q (two 16-row tiles) held for the whole key loop.
  const bf16* qr0 = qp + (((size_t)bh * SEQ + qbase + l15) << 6);
  const bf16* qr1 = qr0 + (16 << 6);
  bf16x16 aq[2][2];
  aq[0][0] = ld_frag(qr0 + klo,      qr0 + klo + 8);
  aq[0][1] = ld_frag(qr0 + 32 + klo, qr0 + 32 + klo + 8);
  aq[1][0] = ld_frag(qr1 + klo,      qr1 + klo + 8);
  aq[1][1] = ld_frag(qr1 + 32 + klo, qr1 + 32 + klo + 8);

  const bf16*  kb = kp + (((size_t)bh * SEQ) << 6);
  const bf16*  vb = vt + (size_t)bh * HD * SEQ;
  const float* Mb = Mbuf + (size_t)bh * SEQ;
  const float* Lb = Linv + (size_t)bh * SEQ;

  const int tid = threadIdx.x;
  const int krow = tid >> 3;        // 0..31 for K staging
  const int kcol = (tid & 7) * 8;   // 16B chunk within 64-elem K row
  const int vrow = tid >> 2;        // 0..63 (only <48 used) for V staging
  const int vcol = (tid & 3) * 8;   // 16B chunk within 32-elem V row

  f32x8 acc[2][3] = {{f32x8{}, f32x8{}, f32x8{}}, {f32x8{}, f32x8{}, f32x8{}}};

  for (int kc = 0; kc < SEQ / 32; ++kc) {
    // Stage shared K (32x64) and V (48x32) tiles for this 32-key chunk.
    stage16(kb + ((size_t)(kc * 32 + krow) << 6) + kcol, &kstage[krow][kcol]);
    if (tid < 192)
      stage16(vb + (size_t)vrow * SEQ + kc * 32 + vcol, &vstage[vrow][vcol]);
    stage_wait();
    __syncthreads();

#pragma unroll
    for (int half = 0; half < 2; ++half) {
      const int kt = kc * 32 + half * 16;
      const bf16* krl = &kstage[half * 16 + l15][0];
      bf16x16 kf0 = ld_frag(krl + klo,      krl + klo + 8);
      bf16x16 kf1 = ld_frag(krl + 32 + klo, krl + 32 + klo + 8);
      const float mv = Mb[kt + l15];
      const float li = Lb[kt + l15];
#pragma unroll
      for (int t2 = 0; t2 < 2; ++t2) {
        f32x8 s = {};
        s = wmma_bf16(aq[t2][0], kf0, s);
        s = wmma_bf16(aq[t2][1], kf1, s);
#pragma unroll
        for (int r = 0; r < 8; ++r) {
          float p = __expf(s[r] - mv) * li;
          pstage[wave][t2 * 16 + r + hi8][half * 16 + l15] = (bf16)p;
        }
      }
    }
    // pstage is wave-private; DS ops are in-order per wave, so only a
    // compiler barrier is needed before re-reading as A-fragments.
    asm volatile("" ::: "memory");

#pragma unroll
    for (int t2 = 0; t2 < 2; ++t2) {
      const bf16* prl = &pstage[wave][t2 * 16 + l15][0];
      bf16x16 pf = ld_frag(prl + klo, prl + klo + 8);
#pragma unroll
      for (int nt = 0; nt < 3; ++nt) {
        const bf16* vrl = &vstage[nt * 16 + l15][0];
        bf16x16 vf = ld_frag(vrl + klo, vrl + klo + 8);
        acc[t2][nt] = wmma_bf16(pf, vf, acc[t2][nt]);
      }
    }
    __syncthreads();   // protect kstage/vstage/pstage before next chunk
  }

#pragma unroll
  for (int t2 = 0; t2 < 2; ++t2) {
#pragma unroll
    for (int nt = 0; nt < 3; ++nt) {
      const int col = h * HD + nt * 16 + l15;
#pragma unroll
      for (int r = 0; r < 8; ++r) {
        int q = qbase + t2 * 16 + r + hi8;
        attnC[((size_t)b * SEQ + q) * DIMK + col] = (bf16)acc[t2][nt][r];
      }
    }
  }
}

// ---------------------------------------------------------------------------
// Launch
// ---------------------------------------------------------------------------
extern "C" void kernel_launch(void* const* d_in, const int* in_sizes, int n_in,
                              void* d_out, int out_size, void* d_ws, size_t ws_size,
                              hipStream_t stream) {
  (void)in_sizes; (void)n_in; (void)out_size; (void)ws_size;
  const float* x  = (const float*)d_in[0];
  const float* Wq = (const float*)d_in[1];
  const float* bq = (const float*)d_in[2];   // [H,D] flat == bias[h*48+d]
  const float* Wk = (const float*)d_in[3];
  const float* bk = (const float*)d_in[4];
  const float* Wv = (const float*)d_in[5];
  const float* bv = (const float*)d_in[6];
  const float* Wo = (const float*)d_in[7];
  const float* bo = (const float*)d_in[8];
  float* out = (float*)d_out;

  // Workspace layout (bytes, all 16B aligned). Total ~73.5 MB.
  char* ws = (char*)d_ws;
  constexpr size_t XB_B  = (size_t)NROWS * DIMK * 2;       // 12,582,912
  constexpr size_t WT_B  = (size_t)DIMK * DIMK * 2;        //  1,179,648
  constexpr size_t QK_B  = (size_t)BH * SEQ * HDP * 2;     // 16,777,216
  constexpr size_t VT_B  = (size_t)BH * HD * SEQ * 2;      // 12,582,912
  constexpr size_t AC_B  = XB_B;                           // 12,582,912
  constexpr size_t ML_B  = (size_t)BH * SEQ * 4;           //    524,288

  bf16*  xb    = (bf16*)(ws);
  bf16*  wqt   = (bf16*)(ws + XB_B);
  bf16*  wkt   = (bf16*)(ws + XB_B + WT_B);
  bf16*  wvt   = (bf16*)(ws + XB_B + 2 * WT_B);
  bf16*  wot   = (bf16*)(ws + XB_B + 3 * WT_B);
  bf16*  qpad  = (bf16*)(ws + XB_B + 4 * WT_B);
  bf16*  kpad  = (bf16*)(ws + XB_B + 4 * WT_B + QK_B);
  bf16*  vtw   = (bf16*)(ws + XB_B + 4 * WT_B + 2 * QK_B);
  bf16*  attnC = (bf16*)(ws + XB_B + 4 * WT_B + 2 * QK_B + VT_B);
  float* Mbuf  = (float*)(ws + XB_B + 4 * WT_B + 2 * QK_B + VT_B + AC_B);
  float* Linv  = (float*)(ws + XB_B + 4 * WT_B + 2 * QK_B + VT_B + AC_B + ML_B);

  // 1) zero padded Q/K buffers (pad lanes d=48..63 must be 0)
  {
    int n_u4 = (int)((2 * QK_B) / 16);
    zero_u4_kernel<<<(n_u4 + 255) / 256, 256, 0, stream>>>((uint4*)qpad, n_u4);
  }
  // 2) convert x -> bf16
  {
    int n4 = NROWS * DIMK / 4;
    cvt_x_kernel<<<(n4 + 255) / 256, 256, 0, stream>>>(x, xb, n4);
  }
  // 3) weight transposes -> bf16 B^T layouts
  {
    int nw = NH * DIMK * HD;  // 589,824
    build_wht_kernel<<<(nw + 255) / 256, 256, 0, stream>>>(Wq, wqt, nw);
    build_wht_kernel<<<(nw + 255) / 256, 256, 0, stream>>>(Wk, wkt, nw);
    build_wht_kernel<<<(nw + 255) / 256, 256, 0, stream>>>(Wv, wvt, nw);
    int no = DIMK * DIMK;
    build_wot_kernel<<<(no + 255) / 256, 256, 0, stream>>>(Wo, wot, no);
  }
  // 4) QKV projection GEMMs (384 blocks of 8 waves: 256 M-tiles x 12 N-groups)
  constexpr int GEMM_BLOCKS = (NROWS / 32) * (DIMK / 64) / 8;  // 384
  gemm_kernel<<<GEMM_BLOCKS, 256, 0, stream>>>(xb, wqt, bq, qpad, nullptr, 0, kInvSqrtD);
  gemm_kernel<<<GEMM_BLOCKS, 256, 0, stream>>>(xb, wkt, bk, kpad, nullptr, 1, 1.0f);
  gemm_kernel<<<GEMM_BLOCKS, 256, 0, stream>>>(xb, wvt, bv, vtw,  nullptr, 2, 1.0f);
  // 5) column-softmax stats over the query axis
  softmax_stats_kernel<<<BH * 4, 256, 0, stream>>>(qpad, kpad, Mbuf, Linv);
  // 6) p @ V with score recompute
  attn_pv_kernel<<<BH * 4, 256, 0, stream>>>(qpad, kpad, vtw, Mbuf, Linv, attnC);
  // 7) output projection -> fp32 d_out
  gemm_kernel<<<BH * 3, 256, 0, stream>>>(attnC, wot, bo, nullptr, out, 3, 1.0f);
}
